// gru_63591285785309
// MI455X (gfx1250) — matrix operand order repaired and checked by reading, not compiled, not run on hardware
//
#include <hip/hip_runtime.h>

// GRU scan B=128, S=512, H=1024 on MI455X (gfx1250).
// Per step: r = sig(p_xi@Ww^T+Wb); h = tanh((r*p+x)@Vw^T+Vb); z = sig(p_xi@Uw^T+Ub);
//           p' = (1-z)p + z h,  p_xi = x + p.
//
// All GEMMs run as v_wmma_f32_16x16x32_bf16 with fp32 accumulate.
// bf16 operands are produced ONCE outside the GEMM loops:
//   - weights converted once per launch (6 MB, L2-resident)
//   - p_xi(t+1) emitted in bf16 by step-t's epilogue (init kernel seeds t=0)
//   - a = r*p+x emitted in bf16 by the gates epilogue
// GEMM inner loops are pure global_load_b128 + prefetch + WMMA: no LDS,
// no barriers, no converts, no branches.

typedef __attribute__((ext_vector_type(16))) __bf16 v16bf;
typedef __attribute__((ext_vector_type(8)))  float  v8f;

union Frag { v16bf v; uint4 q[2]; };

#define BB   128
#define SS   512
#define HH   1024
#define PSTR (SS * HH)   // stride (floats) between batches within one time-slice of emb/out

#define WMMA_BF16(A, B, C) \
    __builtin_amdgcn_wmma_f32_16x16x32_bf16(false, (A).v, false, (B).v, (short)0, (C), false, false)

// round-to-nearest-even f32 -> bf16 bits
static __device__ __forceinline__ unsigned short f2bf(float f) {
    unsigned int u = __float_as_uint(f);
    return (unsigned short)((u + 0x7FFFu + ((u >> 16) & 1u)) >> 16);
}

// ---------------------------------------------------------------- one-time: weights -> bf16
__global__ __launch_bounds__(256) void cvt_kernel(
    const float* __restrict__ W, const float* __restrict__ U, const float* __restrict__ V,
    unsigned short* __restrict__ Wo, unsigned short* __restrict__ Uo, unsigned short* __restrict__ Vo)
{
    const int Q = (HH * HH) / 4;
    int i = blockIdx.x * blockDim.x + threadIdx.x;   // 3*Q threads
    const float* src; unsigned short* dst; int r;
    if (i < Q)          { src = W; dst = Wo; r = i; }
    else if (i < 2 * Q) { src = U; dst = Uo; r = i - Q; }
    else                { src = V; dst = Vo; r = i - 2 * Q; }
    float4 f = ((const float4*)src)[r];
    unsigned long long pk =
        (unsigned long long)f2bf(f.x)         |
        ((unsigned long long)f2bf(f.y) << 16) |
        ((unsigned long long)f2bf(f.z) << 32) |
        ((unsigned long long)f2bf(f.w) << 48);
    ((unsigned long long*)dst)[r] = pk;
}

// ---------------------------------------------------------------- one-time: p0 = 0, pxi = bf16(x[:,0,:])
__global__ __launch_bounds__(256) void init_kernel(
    const float* __restrict__ emb, unsigned short* __restrict__ pxi, float* __restrict__ p0)
{
    int i = blockIdx.x * blockDim.x + threadIdx.x;   // (B*H)/4 threads
    ((float4*)p0)[i] = make_float4(0.f, 0.f, 0.f, 0.f);
    int j = i * 4;
    int b = j >> 10, n = j & (HH - 1);
    float4 x = *(const float4*)(emb + (size_t)b * SS * HH + n);   // t = 0 slice
    unsigned long long pk =
        (unsigned long long)f2bf(x.x)         |
        ((unsigned long long)f2bf(x.y) << 16) |
        ((unsigned long long)f2bf(x.z) << 32) |
        ((unsigned long long)f2bf(x.w) << 48);
    ((unsigned long long*)pxi)[i] = pk;
}

// ---------------------------------------------------------------- GEMM core: one 16x16 tile, K=1024
// A fragment (16-bit A layout): lane row = lane&15, K-half offset kb = (lane&16)?8:0,
// elements [0..7] = K kb..kb+7, [8..15] = K kb+16..kb+23 -> two aligned b128 loads.
// B fragment: column lN = W-row lN (y = x@W^T), same K order as A: kb = 8*(lane>>4),
// elements [0..7] = K kb..kb+7, [8..15] = K kb+16..kb+23 -> two aligned b128 loads.
static __device__ __forceinline__ v8f gemm_tile(
    const unsigned short* __restrict__ ap,   // pxi/a row base: + (lane&16?8:0) already applied
    const unsigned short* __restrict__ wp)   // weight row base: + 8*(lane>>4) already applied
{
    v8f acc = {};
    #pragma unroll
    for (int k0 = 0; k0 < HH; k0 += 64) {
        Frag A0, A1, B0, B1;
        A0.q[0] = *(const uint4*)(ap + k0);
        A0.q[1] = *(const uint4*)(ap + k0 + 16);
        A1.q[0] = *(const uint4*)(ap + k0 + 32);
        A1.q[1] = *(const uint4*)(ap + k0 + 48);
        B0.q[0] = *(const uint4*)(wp + k0);
        B0.q[1] = *(const uint4*)(wp + k0 + 16);
        B1.q[0] = *(const uint4*)(wp + k0 + 32);
        B1.q[1] = *(const uint4*)(wp + k0 + 48);
        __builtin_prefetch(wp + k0 + 64, 0, 1);   // next K-slice of weights
        acc = WMMA_BF16(A0, B0, acc);
        acc = WMMA_BF16(A1, B1, acc);
    }
    return acc;
}

// ---------------------------------------------------------------- step kernel A: r & z gates
// grid = 128 blocks x 256 threads; wave = one 16x16 tile of one gate.
// blockIdx.x: mt = >>4 (batch tile), nc = &15 (64-col group); wave w: gate = w>>2, subtile = w&3.
__global__ __launch_bounds__(256) void gates_kernel(
    const unsigned short* __restrict__ pxi,   // [B,H] bf16  (x_t + p_{t-1})
    const unsigned short* __restrict__ Wbf, const float* __restrict__ Wb,
    const unsigned short* __restrict__ Ubf, const float* __restrict__ Ub,
    const float* __restrict__ emb_t,          // x_t: (b,n) at emb_t[b*PSTR + n]
    const float* __restrict__ pprev, size_t pstride,   // p_{t-1}: (b,n) at pprev[b*pstride + n]
    unsigned short* __restrict__ abf,         // [B,H] bf16  a = r*p + x
    float* __restrict__ zbuf)                 // [B,H] f32
{
    const int w    = threadIdx.x >> 5;
    const int lane = threadIdx.x & 31;
    const int mt = blockIdx.x >> 4, nc = blockIdx.x & 15;
    const int g  = w >> 2;
    const int lN    = (nc * 4 + (w & 3)) * 16 + (lane & 15);
    const int bRow0 = mt * 16;
    const int arow  = bRow0 + (lane & 15);
    const int kb    = (lane & 16) ? 8 : 0;

    const unsigned short* ap = pxi + (size_t)arow * HH + kb;
    const unsigned short* wp = (g ? Ubf : Wbf) + (size_t)lN * HH + ((lane >> 4) << 3);

    v8f acc = gemm_tile(ap, wp);

    const float bias = (g ? Ub : Wb)[lN];
    #pragma unroll
    for (int j = 0; j < 8; ++j) {
        int b = bRow0 + j + ((lane & 16) ? 8 : 0);   // C layout: lanes 16-31 hold M+8
        float sg = 1.f / (1.f + __expf(-(acc[j] + bias)));
        if (g == 0) {
            float p = pprev[(size_t)b * pstride + lN];
            float x = emb_t[(size_t)b * PSTR + lN];
            abf[(size_t)b * HH + lN] = f2bf(sg * p + x);
        } else {
            zbuf[(size_t)b * HH + lN] = sg;
        }
    }
}

// ---------------------------------------------------------------- step kernel B: h, p', next p_xi
// grid = 64 blocks x 256 threads; wave = one 16x16 tile.
__global__ __launch_bounds__(256) void out_kernel(
    const unsigned short* __restrict__ abf,
    const unsigned short* __restrict__ Vbf, const float* __restrict__ Vb,
    const float* __restrict__ zbuf,
    const float* __restrict__ pprev, size_t pstride,
    const float* __restrict__ emb_next,       // x_{t+1} slice base, nullptr at last step
    float* __restrict__ out_t,                // ys[:,t,:]: (b,n) at out_t[b*PSTR + n]
    unsigned short* __restrict__ pxi)         // next-step p_xi (bf16)
{
    const int w    = threadIdx.x >> 5;
    const int lane = threadIdx.x & 31;
    const int mt = blockIdx.x >> 3, nc = blockIdx.x & 7;
    const int lN    = (nc * 8 + w) * 16 + (lane & 15);
    const int bRow0 = mt * 16;
    const int arow  = bRow0 + (lane & 15);
    const int kb    = (lane & 16) ? 8 : 0;

    const unsigned short* ap = abf + (size_t)arow * HH + kb;
    const unsigned short* wp = Vbf + (size_t)lN * HH + ((lane >> 4) << 3);

    v8f acc = gemm_tile(ap, wp);

    const float bias = Vb[lN];
    #pragma unroll
    for (int j = 0; j < 8; ++j) {
        int b = bRow0 + j + ((lane & 16) ? 8 : 0);
        float h  = tanhf(acc[j] + bias);
        float z  = zbuf[(size_t)b * HH + lN];
        float p  = pprev[(size_t)b * pstride + lN];
        float pn = (1.f - z) * p + z * h;
        out_t[(size_t)b * PSTR + lN] = pn;                 // p_new == ys[:,t,:]
        if (emb_next)
            pxi[(size_t)b * HH + lN] = f2bf(pn + emb_next[(size_t)b * PSTR + lN]);
    }
}

// ---------------------------------------------------------------- launcher
extern "C" void kernel_launch(void* const* d_in, const int* in_sizes, int n_in,
                              void* d_out, int out_size, void* d_ws, size_t ws_size,
                              hipStream_t stream)
{
    const float* emb = (const float*)d_in[0];
    const float* Ww  = (const float*)d_in[1];
    const float* Wb  = (const float*)d_in[2];
    const float* Uw  = (const float*)d_in[3];
    const float* Ub  = (const float*)d_in[4];
    const float* Vw  = (const float*)d_in[5];
    const float* Vb  = (const float*)d_in[6];
    float* out = (float*)d_out;

    char* ws = (char*)d_ws;
    unsigned short* Wbf = (unsigned short*)(ws);                    // 2 MB
    unsigned short* Ubf = (unsigned short*)(ws + (2ull << 20));     // 2 MB
    unsigned short* Vbf = (unsigned short*)(ws + (4ull << 20));     // 2 MB
    unsigned short* pxi = (unsigned short*)(ws + (6ull << 20));     // 256 KB bf16 [B,H]
    unsigned short* abf = (unsigned short*)(ws + (6ull << 20) + (256ull << 10)); // 256 KB
    float* zbuf = (float*)(ws + (6ull << 20) + (512ull << 10));     // 512 KB
    float* p0   = (float*)(ws + (7ull << 20));                      // 512 KB zeros

    cvt_kernel<<<(3 * HH * HH / 4) / 256, 256, 0, stream>>>(Ww, Uw, Vw, Wbf, Ubf, Vbf);
    init_kernel<<<(BB * HH / 4) / 256, 256, 0, stream>>>(emb, pxi, p0);

    for (int t = 0; t < SS; ++t) {
        const float* pprev  = t ? (out + (size_t)(t - 1) * HH) : p0;
        size_t       pstrid = t ? (size_t)PSTR : (size_t)HH;
        gates_kernel<<<128, 256, 0, stream>>>(pxi, Wbf, Wb, Ubf, Ub,
                                              emb + (size_t)t * HH,
                                              pprev, pstrid, abf, zbuf);
        out_kernel<<<64, 256, 0, stream>>>(abf, Vbf, Vb, zbuf, pprev, pstrid,
                                           (t + 1 < SS) ? (emb + (size_t)(t + 1) * HH) : nullptr,
                                           out + (size_t)t * HH, pxi);
    }
}
